// SpatialCoherenceLoss_36670430773470
// MI455X (gfx1250) — matrix operand order repaired and checked by reading, not compiled, not run on hardware
//
#include <hip/hip_runtime.h>

// SpatialCoherenceLoss for MI455X (gfx1250, wave32, WMMA).
//   out = mean_{i,j}[ exp(-100*||s_i-s_j||^2) * (sqn_i + sqn_j - 2*<p_i,p_j>) ]
// GEMM P*P^T on v_wmma_f32_16x16x32_f16. Spatial coords pre-scaled by
// sqrt(100*log2(e)) so the weight is exp2(-(dx^2+dy^2)): the -100 and log2e
// multiplies vanish from the N^2 epilogue and the negate folds into
// v_exp_f32's source modifier. Deterministic two-stage reduction.

typedef __attribute__((ext_vector_type(16))) _Float16 v16h;
typedef __attribute__((ext_vector_type(8)))  _Float16 v8h;
typedef __attribute__((ext_vector_type(2)))  _Float16 v2h;
typedef __attribute__((ext_vector_type(8)))  float    v8f;

#define NPTS 8192
#define DIM  64
#define TILE 16
#define NT   (NPTS / TILE)            // 512 tiles per matrix dim
#define WPB  8                        // waves per block (256 threads)
#define JTPW 4                        // j-tiles per wave
#define JTPB (WPB * JTPW)             // 32 j-tiles per block
#define GY   (NT / JTPB)              // 16 blocks in y
#define NPART (NT * GY)               // 8192 block partials

// sqrt(100 * log2(e)) ; weight = exp(-100*d2) = exp2(-(s*dx)^2 - (s*dy)^2)
#define COORD_SCALE 12.011224087864498f

union Frag { v16h v; v8h h[2]; };

// --- Pass 1: P f32->f16, per-row squared norms, pre-scaled spatial coords.
__global__ void prep_kernel(const float* __restrict__ pred,
                            const float* __restrict__ sc,
                            _Float16* __restrict__ ph,
                            float* __restrict__ sqn,
                            float* __restrict__ scs) {
    const int lane = threadIdx.x & 31;
    const int row  = blockIdx.x * WPB + (threadIdx.x >> 5);
    const float* r = pred + (size_t)row * DIM;
    float a = r[lane * 2 + 0];
    float b = r[lane * 2 + 1];
    v2h hv; hv[0] = (_Float16)a; hv[1] = (_Float16)b;
    *(v2h*)(ph + (size_t)row * DIM + lane * 2) = hv;
    if (lane < 2)
        scs[row * 2 + lane] = sc[row * 2 + lane] * COORD_SCALE;
    float s = a * a + b * b;
    #pragma unroll
    for (int off = 16; off > 0; off >>= 1) s += __shfl_xor(s, off, 32);
    if (lane == 0) sqn[row] = s;
}

// --- Pass 2: tiled pairwise loss. One 16x16 output tile per wave per step.
__global__ void __launch_bounds__(256)
loss_kernel(const _Float16* __restrict__ ph,
            const float* __restrict__ sqn,
            const float* __restrict__ scs,     // [NPTS,2] pre-scaled coords
            float* __restrict__ partial) {
    __shared__ float red[WPB];
    const int lane = threadIdx.x & 31;
    const int wid  = threadIdx.x >> 5;

    const int i_base = blockIdx.x * TILE;
    const int m  = lane & 15;               // row (A) / col (B) within tile
    const int hb = lane >> 4;               // lane half

    // A fragment (16x32 f16 per K-chunk): lane-half0 K{0..7,16..23},
    // half1 K{8..15,24..31}  ->  two 16B loads per chunk at offsets kb, kb+16.
    const _Float16* arow = ph + (size_t)(i_base + m) * DIM;
    const int kb = hb << 3;                 // 0 or 8
    Frag a0, a1;
    a0.h[0] = *(const v8h*)(arow + kb);
    a0.h[1] = *(const v8h*)(arow + kb + 16);
    a1.h[0] = *(const v8h*)(arow + 32 + kb);
    a1.h[1] = *(const v8h*)(arow + 32 + kb + 16);

    // Epilogue i-side data: this lane covers output rows i_base+mb+v, v=0..7.
    const int mb = hb << 3;
    float sqi[8], sxi[8], syi[8];
    #pragma unroll
    for (int v = 0; v < 8; ++v) {
        const int r = i_base + mb + v;
        sqi[v] = sqn[r];
        sxi[v] = scs[r * 2 + 0];
        syi[v] = scs[r * 2 + 1];
    }

    float lsum = 0.0f;

    #pragma unroll
    for (int t = 0; t < JTPW; ++t) {
        const int jt     = blockIdx.y * JTPB + wid * JTPW + t;
        const int j_base = jt * TILE;

        // B fragment (32x16 f16 per K-chunk): lane-half0 K{0..7,16..23},
        // half1 K{8..15,24..31}  ->  two 16B loads at offsets ko, ko+16.
        const _Float16* brow = ph + (size_t)(j_base + m) * DIM;
        const int ko = hb << 3;             // 0 or 8
        Frag b0, b1;
        b0.h[0] = *(const v8h*)(brow + ko);
        b0.h[1] = *(const v8h*)(brow + ko + 16);
        b1.h[0] = *(const v8h*)(brow + 32 + ko);
        b1.h[1] = *(const v8h*)(brow + 32 + ko + 16);

        // dot[i,j] over K=64 in two K=32 WMMA steps.
        v8f acc = {};
        acc = __builtin_amdgcn_wmma_f32_16x16x32_f16(
                  false, a0.v, false, b0.v, (short)0, acc, false, false);
        acc = __builtin_amdgcn_wmma_f32_16x16x32_f16(
                  false, a1.v, false, b1.v, (short)0, acc, false, false);

        // Epilogue j-side data: this lane owns output column j_base + (lane&15).
        const int c   = j_base + m;
        const float sqj = sqn[c];
        const float sxj = scs[c * 2 + 0];
        const float syj = scs[c * 2 + 1];

        #pragma unroll
        for (int v = 0; v < 8; ++v) {
            const float pd = sqi[v] + sqj - 2.0f * acc[v];
            const float dx = sxi[v] - sxj;
            const float dy = syi[v] - syj;
            // weight = exp2(-(dx^2+dy^2)); negate folds into v_exp_f32 src mod
            const float w  = __builtin_amdgcn_exp2f(-(dx * dx + dy * dy));
            lsum = fmaf(w, pd, lsum);
        }
    }

    // wave32 reduce, then block reduce, one partial per block (deterministic).
    #pragma unroll
    for (int off = 16; off > 0; off >>= 1) lsum += __shfl_xor(lsum, off, 32);
    if (lane == 0) red[wid] = lsum;
    __syncthreads();
    if (threadIdx.x == 0) {
        float s = 0.0f;
        #pragma unroll
        for (int wv = 0; wv < WPB; ++wv) s += red[wv];
        partial[blockIdx.y * gridDim.x + blockIdx.x] = s;
    }
}

// --- Pass 3: fixed-order reduction of block partials -> scalar mean.
__global__ void reduce_kernel(const float* __restrict__ partial,
                              float* __restrict__ out) {
    __shared__ float red[256];
    float s = 0.0f;
    for (int i = threadIdx.x; i < NPART; i += 256) s += partial[i];
    red[threadIdx.x] = s;
    __syncthreads();
    #pragma unroll
    for (int off = 128; off > 0; off >>= 1) {
        if ((int)threadIdx.x < off) red[threadIdx.x] += red[threadIdx.x + off];
        __syncthreads();
    }
    if (threadIdx.x == 0)
        out[0] = red[0] * (1.0f / ((float)NPTS * (float)NPTS));
}

extern "C" void kernel_launch(void* const* d_in, const int* in_sizes, int n_in,
                              void* d_out, int out_size, void* d_ws, size_t ws_size,
                              hipStream_t stream) {
    const float* pred = (const float*)d_in[0];   // [8192, 64] f32
    const float* sc   = (const float*)d_in[1];   // [8192, 2]  f32
    float* out = (float*)d_out;                  // scalar f32

    // Workspace: f16 P (1 MB) | sqnorms (32 KB) | partials (32 KB) | scaled coords (64 KB)
    _Float16* ph   = (_Float16*)d_ws;
    float* sqn     = (float*)((char*)d_ws + (size_t)NPTS * DIM * sizeof(_Float16));
    float* partial = sqn + NPTS;
    float* scs     = partial + NPART;

    prep_kernel<<<NPTS / WPB, 256, 0, stream>>>(pred, sc, ph, sqn, scs);
    loss_kernel<<<dim3(NT, GY), 256, 0, stream>>>(ph, sqn, scs, partial);
    reduce_kernel<<<1, 256, 0, stream>>>(partial, out);
}
